// RNN_57939108823393
// MI455X (gfx1250) — matrix operand order repaired and checked by reading, not compiled, not run on hardware
//
#include <hip/hip_runtime.h>

// ---------------------------------------------------------------------------
// RNN on MI455X (gfx1250): bf16 WMMA GEMMs + fused LeakyReLU epilogue.
//   h_new_all = leaky(x @ W_in + b_in)                 (one big GEMM)
//   h_{t+1}   = leaky([h_new_t, h_t] @ W_h + b_h)      (1024 small GEMMs)
//   out       = h_final @ W_out + b_out                (one small GEMM)
// bf16 matmuls, fp32 accumulate (v_wmma_f32_16x16x32_bf16).
// A-tiles are double-buffered in LDS and filled with the CDNA5 async path
// (global_load_async_to_lds_b128 + s_wait_asynccnt), one barrier per K-step.
// ---------------------------------------------------------------------------

typedef __bf16 bf16;
typedef __attribute__((ext_vector_type(16))) __bf16 v16bf;
typedef __attribute__((ext_vector_type(8)))  float  v8f;

static constexpr int BB = 64;    // batch
static constexpr int TT = 1024;  // time steps
static constexpr int II = 512;   // input dim
static constexpr int HH = 1024;  // hidden dim
static constexpr int OO = 512;   // output dim

// ------------------------------ elementwise --------------------------------
__global__ void k_cvt_bf16(const float* __restrict__ s, bf16* __restrict__ d, int n) {
    int i = blockIdx.x * blockDim.x + threadIdx.x;
    if (i < n) d[i] = (bf16)s[i];
}

__global__ void k_zero_bf16(bf16* __restrict__ p, int n) {
    int i = blockIdx.x * blockDim.x + threadIdx.x;
    if (i < n) p[i] = (bf16)0.0f;
}

// ------------------------- weight fragment swizzle -------------------------
// Pack row-major fp32 W[K][N] into bf16 WMMA B-fragments.
// Fragment (ntile, kt) covers N cols [ntile*16,+16), K rows [kt*32,+32).
// Lane layout this code uses for the V_WMMA_F32_16X16X32_BF16 B operand (wave32, contiguous k):
//   lane l: n = (l & 15), khalf = l >> 4, element e (0..15) -> K = khalf*16 + e.
// Stored as two 16-byte chunks per lane, lane-major per chunk, so the GEMM
// inner loop does two fully-coalesced global_load_b128 per fragment.
__global__ void k_swz_w(const float* __restrict__ W, bf16* __restrict__ dst,
                        int K, int N) {
    int tid = blockIdx.x * blockDim.x + threadIdx.x;
    int total = (N / 16) * (K / 32) * 64;       // 2 chunks * 32 lanes per frag
    if (tid >= total) return;
    int lane  = tid & 31;
    int c     = (tid >> 5) & 1;
    int frag  = tid >> 6;
    int nKt   = K / 32;
    int kt    = frag % nKt;
    int ntile = frag / nKt;
    int khalf = lane >> 4;
    int n     = ntile * 16 + (lane & 15);
    int kbase = kt * 32 + khalf * 16 + c * 8;
    alignas(16) bf16 tmp[8];
#pragma unroll
    for (int i = 0; i < 8; ++i)
        tmp[i] = (bf16)W[(long)(kbase + i) * N + n];
    *(uint4*)(dst + ((size_t)frag * 2 + c) * 256 + lane * 8) = *(const uint4*)tmp;
}

// ------------------------------- WMMA GEMM ---------------------------------
// C[M,N] = act(A @ B + bias).  A bf16 row-major, split at Ksplit between two
// sources (the [h_new_t, h_prev] concat); B pre-swizzled bf16 fragments;
// bias fp32; output bf16 or fp32.
// Block: 256 threads = 8 waves; tile 64(M) x 128(N); wave tile 16 x 64.
template <bool LEAKY, bool OUTBF16>
__global__ void k_gemm(const bf16* __restrict__ A1, long lda1,
                       const bf16* __restrict__ A2, long lda2, int Ksplit,
                       const bf16* __restrict__ Bsw,
                       const float* __restrict__ bias,
                       void* __restrict__ Cout, long ldc, int K) {
    // Double-buffered 64x32 bf16 A tile; rows padded to 40 elems (80 B).
    __shared__ bf16 ldsA[2 * 64 * 40];

    const int tid   = threadIdx.x;
    const int lane  = tid & 31;
    const int wave  = tid >> 5;
    const int wm    = wave & 3;       // 4 waves along M -> 16 rows each
    const int wn    = wave >> 2;      // 2 waves along N -> 64 cols each
    const int khalf = lane >> 4;
    const long m0   = (long)blockIdx.y * 64;
    const int  n0   = blockIdx.x * 128;
    const int  nKt  = K / 32;

    // A-tile staging: thread -> (row, K-block-of-8).  LDS row K order is
    // [K0-7 | K8-15 | K16-23 | K24-31]: lane khalf=0 reads elems 0..15
    // (K 0-15), khalf=1 reads elems 16..31 (K 16-31), each one
    // contiguous 32-byte chunk matching the B-fragment element order.
    const int arow  = tid >> 2;                       // 0..63
    const int kb    = tid & 3;                        // K block of 8
    const int aslot = kb;                             // 0,1,2,3 -> 0,1,2,3
    const int kbOff = kb * 8;

    const long  rowg  = m0 + arow;
    const bf16* aSrc1 = A1 + rowg * lda1 + kbOff;            // valid k < Ksplit
    const bf16* aSrc2 = A2 + rowg * lda2 + kbOff - Ksplit;   // valid k >= Ksplit
    // LDS byte address for async dst (generic LDS pointer keeps the offset in
    // its low 32 bits; async dsaddr is LDS_BASE-relative like DS addressing).
    const unsigned ldsDstBase = (unsigned)(uintptr_t)(void*)ldsA
                              + (unsigned)(arow * 80 + aslot * 16);

    // Per-lane B fragment chunk-0 pointers (chunk1 at +256 elems).
    const bf16* bp[4];
#pragma unroll
    for (int i = 0; i < 4; ++i) {
        int ntile = (n0 + wn * 64 + i * 16) >> 4;
        bp[i] = Bsw + (size_t)ntile * nKt * 512 + lane * 8;
    }

    // A-fragment LDS read pointer (buffer 0); buffer 1 at +2560 elems.
    const bf16* afp = ldsA + (size_t)(wm * 16 + (lane & 15)) * 40 + khalf * 16;

    v8f acc[4] = {v8f{}, v8f{}, v8f{}, v8f{}};

    // Async prefetch of A tile `ktn` into LDS buffer (ktn & 1).
    auto prefetchA = [&](int ktn) {
        int kg = ktn * 32 + kbOff;
        const bf16* src = (kg < Ksplit) ? (aSrc1 + (size_t)ktn * 32)
                                        : (aSrc2 + (size_t)ktn * 32);
        unsigned dst = ldsDstBase + (unsigned)((ktn & 1) * 5120);
        asm volatile("global_load_async_to_lds_b128 %0, %1, off"
                     :: "v"(dst), "v"(src) : "memory");
    };

    prefetchA(0);

    for (int kt = 0; kt < nKt; ++kt) {
        // Tile (kt) landed in LDS for this wave; barrier makes it visible to
        // all waves and proves buffer (kt+1)&1 has been fully consumed.
        asm volatile("s_wait_asynccnt 0x0" ::: "memory");
        __syncthreads();
        if (kt + 1 < nKt) prefetchA(kt + 1);

        // A fragment: one lane = one row, 32 contiguous LDS bytes.
        const bf16* ab = afp + (kt & 1) * 2560;
        union { uint4 u[2]; v16bf v; } af;
        af.u[0] = *(const uint4*)(ab);
        af.u[1] = *(const uint4*)(ab + 8);

        // Issue all B-fragment loads first (one clause, one wait), then WMMAs.
        union { uint4 u[2]; v16bf v; } bfg[4];
#pragma unroll
        for (int i = 0; i < 4; ++i) {
            const bf16* p = bp[i] + (size_t)kt * 512;
            bfg[i].u[0] = *(const uint4*)(p);
            bfg[i].u[1] = *(const uint4*)(p + 256);
        }
#pragma unroll
        for (int i = 0; i < 4; ++i)
            acc[i] = __builtin_amdgcn_wmma_f32_16x16x32_bf16(
                false, af.v, false, bfg[i].v, (short)0, acc[i], false, false);
    }

    // Epilogue: C/D 16x16 f32 layout -> lane gives N, VGPR index gives M.
#pragma unroll
    for (int i = 0; i < 4; ++i) {
        int   n  = n0 + wn * 64 + i * 16 + (lane & 15);
        float bv = bias[n];
#pragma unroll
        for (int v = 0; v < 8; ++v) {
            long  m   = m0 + wm * 16 + v + khalf * 8;
            float val = acc[i][v] + bv;
            if (LEAKY) val = (val >= 0.0f) ? val : 0.01f * val;
            if (OUTBF16) ((bf16*)Cout)[m * ldc + n] = (bf16)val;
            else         ((float*)Cout)[m * ldc + n] = val;
        }
    }
}

// ------------------------------- launcher ----------------------------------
extern "C" void kernel_launch(void* const* d_in, const int* in_sizes, int n_in,
                              void* d_out, int out_size, void* d_ws, size_t ws_size,
                              hipStream_t stream) {
    const float* x     = (const float*)d_in[0];
    const float* W_in  = (const float*)d_in[1];
    const float* b_in  = (const float*)d_in[2];
    const float* W_h   = (const float*)d_in[3];
    const float* b_h   = (const float*)d_in[4];
    const float* W_out = (const float*)d_in[5];
    const float* b_out = (const float*)d_in[6];

    // Workspace layout (bf16 elements), ~199 MB total.
    bf16* xb    = (bf16*)d_ws;                          //  64 MB: x in bf16
    bf16* WinS  = xb    + (size_t)BB * TT * II;         //   1 MB
    bf16* WhS   = WinS  + (size_t)II * HH;              //   4 MB
    bf16* WoutS = WhS   + (size_t)2 * HH * HH;          //   1 MB
    bf16* hnew  = WoutS + (size_t)HH * OO;              // 128 MB: leaky(xW_in+b)
    bf16* h0    = hnew  + (size_t)BB * TT * HH;         // 128 KB ping
    bf16* h1    = h0    + (size_t)BB * HH;              // 128 KB pong

    // 1) convert x to bf16; swizzle weights into WMMA B-fragment layout.
    int nx = BB * TT * II;
    k_cvt_bf16<<<(nx + 255) / 256, 256, 0, stream>>>(x, xb, nx);
    {
        int t1 = (HH / 16) * (II / 32) * 64;
        k_swz_w<<<(t1 + 255) / 256, 256, 0, stream>>>(W_in, WinS, II, HH);
        int t2 = (HH / 16) * ((2 * HH) / 32) * 64;
        k_swz_w<<<(t2 + 255) / 256, 256, 0, stream>>>(W_h, WhS, 2 * HH, HH);
        int t3 = (OO / 16) * (HH / 32) * 64;
        k_swz_w<<<(t3 + 255) / 256, 256, 0, stream>>>(W_out, WoutS, HH, OO);
    }
    k_zero_bf16<<<(BB * HH + 255) / 256, 256, 0, stream>>>(h0, BB * HH);

    // 2) Input projection: [B*T, I] @ [I, H] -> bf16 hnew, fused leaky+bias.
    {
        dim3 grid(HH / 128, (BB * TT) / 64);
        k_gemm<true, true><<<grid, 256, 0, stream>>>(
            xb, (long)II, xb, (long)II, II, WinS, b_in, hnew, (long)HH, II);
    }

    // 3) Recurrent scan: 1024 dependent steps of [64, 2H] @ [2H, H].
    //    A is the concat [h_new_t | h_prev] via split-K addressing; W_h (4 MB
    //    bf16) stays L2-resident across all steps.
    bf16* hp = h0;
    bf16* hn = h1;
    for (int t = 0; t < TT; ++t) {
        dim3 grid(HH / 128, BB / 64);
        k_gemm<true, true><<<grid, 256, 0, stream>>>(
            hnew + (size_t)t * HH, (long)TT * HH,   // h_new_t rows, stride T*H
            hp, (long)HH, HH,                       // h_prev for k >= H
            WhS, b_h, hn, (long)HH, 2 * HH);
        bf16* tmp = hp; hp = hn; hn = tmp;
    }

    // 4) Output head: [64, H] @ [H, O] -> fp32 d_out, bias only.
    {
        dim3 grid(OO / 128, BB / 64);
        k_gemm<false, false><<<grid, 256, 0, stream>>>(
            hp, (long)HH, hp, (long)HH, HH, WoutS, b_out, d_out, (long)OO, HH);
    }
}
